// QuantumInspiredAttention_38568806318298
// MI455X (gfx1250) — matrix operand order repaired and checked by reading, not compiled, not run on hardware
//
#include <hip/hip_runtime.h>

// ---------------------------------------------------------------------------
// QuantumInspiredAttention forward for MI455X (gfx1250, wave32, WMMA + TDM)
// ---------------------------------------------------------------------------

typedef __attribute__((ext_vector_type(16))) _Float16 v16h;
typedef __attribute__((ext_vector_type(8)))  _Float16 v8h;
typedef __attribute__((ext_vector_type(4)))  _Float16 v4h;
typedef __attribute__((ext_vector_type(8)))  float    v8f;
typedef __attribute__((ext_vector_type(4)))  unsigned u32x4;
typedef __attribute__((ext_vector_type(8)))  unsigned u32x8;

#define D_MODEL   1024
#define NUM_HEADS 16
#define HEAD_DIM  64
#define BATCH     2
#define SEQ       2048
#define NTOK      (BATCH * SEQ)   // 4096 tokens

// ---------------------------------------------------------------------------
// WMMA helpers (CDNA5 v_wmma_f32_16x16x32_f16, wave32)
// ---------------------------------------------------------------------------
__device__ inline v8f wmma_f16(v16h a, v16h b, v8f c) {
  return __builtin_amdgcn_wmma_f32_16x16x32_f16(false, a, false, b, (short)0, c,
                                                false, false);
}

// A fragment: 16x32 (MxK) f16 row-major tile. ISA: lane(0..15) row=lane,
// chunks K=[0..7],[16..23]; lane(16..31) same row, chunks K=[8..15],[24..31].
__device__ inline v16h load_a_frag(const _Float16* base, int ld) {
  int lane = threadIdx.x & 31;
  const _Float16* p = base + (lane & 15) * ld + ((lane >> 4) << 3);
  v8h lo = *(const v8h*)p;
  v8h hi = *(const v8h*)(p + 16);
  v16h a;
#pragma unroll
  for (int i = 0; i < 8; ++i) { a[i] = lo[i]; a[i + 8] = hi[i]; }
  return a;
}

// B fragment: 32x16 (KxN) with B[k][n] = Nrow[n][k]; `base` = row-major
// [16 x K] tile of N-rows. Lane(0..15) K=[0..7],[16..23]; lane(16..31) K=[8..15],[24..31].
__device__ inline v16h load_b_frag(const _Float16* base, int ld) {
  int lane = threadIdx.x & 31;
  const _Float16* p = base + (lane & 15) * ld + ((lane >> 4) << 3);
  v8h lo = *(const v8h*)p;
  v8h hi = *(const v8h*)(p + 16);
  v16h b;
#pragma unroll
  for (int i = 0; i < 8; ++i) { b[i] = lo[i]; b[i + 8] = hi[i]; }
  return b;
}

// ---------------------------------------------------------------------------
// Prep kernels: fp32 -> f16 conversion + algebraic weight folding
// ---------------------------------------------------------------------------
__global__ __launch_bounds__(256) void cvt_f16(const float* __restrict__ in,
                                               _Float16* __restrict__ out, int n) {
  int i = blockIdx.x * 256 + threadIdx.x;
  if (i < n) out[i] = (_Float16)in[i];
}

// Wq_eff[n,c] = Wq[n,c] * sigmoid(gate[h,d]) / sqrt(HEAD_DIM),  n = h*64+d
__global__ __launch_bounds__(256) void prep_wq(const float* __restrict__ Wq,
                                               const float* __restrict__ gate,
                                               _Float16* __restrict__ out) {
  int i = blockIdx.x * 256 + threadIdx.x;          // 1M elements
  int nrow = i >> 10;
  float g = gate[(nrow >> 6) * 64 + (nrow & 63)];
  float s = 0.125f / (1.0f + __expf(-g));
  out[i] = (_Float16)(Wq[i] * s);
}

__global__ __launch_bounds__(256) void prep_bq(const float* __restrict__ bq,
                                               const float* __restrict__ gate,
                                               float* __restrict__ out) {
  int i = blockIdx.x * 256 + threadIdx.x;
  if (i >= D_MODEL) return;
  float g = gate[(i >> 6) * 64 + (i & 63)];
  out[i] = bq[i] * (0.125f / (1.0f + __expf(-g)));
}

// Wk_eff[h*64+e, c] = sum_d Wk[h*64+d, c] * ent[h,d,e]
__global__ __launch_bounds__(256) void prep_wk(const float* __restrict__ Wk,
                                               const float* __restrict__ ent,
                                               _Float16* __restrict__ out) {
  int i = blockIdx.x * 256 + threadIdx.x;          // 1M elements
  int nrow = i >> 10, c = i & 1023;
  int h = nrow >> 6, e = nrow & 63;
  float s = 0.0f;
#pragma unroll 8
  for (int d = 0; d < 64; ++d)
    s += Wk[(size_t)((h << 6) + d) * 1024 + c] * ent[(((h << 6) + d) << 6) + e];
  out[i] = (_Float16)s;
}

__global__ __launch_bounds__(256) void prep_bk(const float* __restrict__ bk,
                                               const float* __restrict__ ent,
                                               float* __restrict__ out) {
  int i = blockIdx.x * 256 + threadIdx.x;
  if (i >= D_MODEL) return;
  int h = i >> 6, e = i & 63;
  float s = 0.0f;
  for (int d = 0; d < 64; ++d)
    s += bk[(h << 6) + d] * ent[(((h << 6) + d) << 6) + e];
  out[i] = s;
}

// ---------------------------------------------------------------------------
// Tiled WMMA GEMM:  C[4096,1024] = A[4096,1024] @ W[1024,1024]^T + bias
// Block tile 128x128, 8 waves (2x4), wave tile 64x32, K-step 32,
// double-buffered LDS. A-tile staged with per-lane async copies (ASYNCcnt),
// B-tile staged with one Tensor Data Mover descriptor (TENSORcnt) whose LDS
// pad feature (16DW interval / 4DW pad) produces the ld=40 padded layout.
// MODE 0: f16 out [b][h][s][64] (Q/K)   MODE 1: f16 out [b][h][64][s] (V^T)
// MODE 2: f32 out [m][1024] (final projection)
// ---------------------------------------------------------------------------
template <int MODE>
__global__ __launch_bounds__(256) void gemm_wmma(const _Float16* __restrict__ A,
                                                 const _Float16* __restrict__ W,
                                                 const float* __restrict__ bias,
                                                 void* __restrict__ out) {
  __shared__ __align__(16) _Float16 sA[2][128 * 40];  // 128 x 32 halves, ld=40
  __shared__ __align__(16) _Float16 sB[2][128 * 40];
  const int K = 1024;
  int m0 = blockIdx.y * 128;
  int n0 = blockIdx.x * 128;
  int t = threadIdx.x;
  int w = t >> 5, lane = t & 31;
  int wr = (w >> 2) * 64;   // wave row offset in block tile
  int wc = (w & 3) * 32;    // wave col offset in block tile
  int r = t >> 1, c = (t & 1) * 16;   // staging coords: 32B per thread

  v8f acc[4][2] = {};

  // A tile: direct memory->LDS async copies, no VGPR staging (ASYNCcnt)
  auto stage_async = [&](int buf, int k0) {
    unsigned long long ga =
        (unsigned long long)(A + (size_t)(m0 + r) * K + k0 + c);
    unsigned la = (unsigned)(size_t)&sA[buf][r * 40 + c];
    asm volatile("global_load_async_to_lds_b128 %0, %1, off"
                 :: "v"(la), "v"(ga) : "memory");
    asm volatile("global_load_async_to_lds_b128 %0, %1, off offset:16"
                 :: "v"(la), "v"(ga) : "memory");
  };

  // B tile: one TDM descriptor per K-step, issued by wave 0 only (TENSORcnt).
  // 2D tile: dim0 = 32 halves along K (contiguous), dim1 = 128 N-rows.
  auto stage_tdm = [&](int buf, int k0) {
    if (w != 0) return;
    unsigned long long ga =
        (unsigned long long)(W + (size_t)n0 * K + k0);
    unsigned la = (unsigned)(size_t)&sB[buf][0];
    u32x4 g0;
    g0[0] = 1u;                                     // count=1, user mode
    g0[1] = la;                                     // lds_addr (bytes)
    g0[2] = (unsigned)ga;                           // global_addr[31:0]
    g0[3] = (unsigned)(ga >> 32) | (2u << 30);      // addr[56:32] | type=2
    u32x8 g1;
    g1[0] = (1u << 16)        // data_size = 2 bytes
          | (1u << 20)        // pad_enable
          | (3u << 22)        // pad_interval: 16 DWORDs (one 64B row)
          | (3u << 25);       // pad_amount: 4 DWORDs -> LDS row stride 80B
    g1[1] = 1024u << 16;      // tensor_dim0 = 1024 (lo16)
    g1[2] = 1024u << 16;      // dim0 hi16=0 | tensor_dim1 = 1024 (lo16)
    g1[3] = 32u << 16;        // dim1 hi16=0 | tile_dim0 = 32
    g1[4] = 128u;             // tile_dim1 = 128, tile_dim2 = 0
    g1[5] = 1024u;            // tensor_dim0_stride = 1024 (lo32)
    g1[6] = 0u;               // stride0 hi | stride1 lo
    g1[7] = 0u;               // stride1 hi
    asm volatile("tensor_load_to_lds %0, %1" :: "s"(g0), "s"(g1) : "memory");
  };

  stage_async(0, 0);
  stage_tdm(0, 0);
  asm volatile("s_wait_asynccnt 0x0" ::: "memory");
  __builtin_amdgcn_s_wait_tensorcnt(0);
  __syncthreads();

  int buf = 0;
  for (int k0 = 0; k0 < K; k0 += 32, buf ^= 1) {
    // prefetch next tiles into the other buffer while we compute this one
    if (k0 + 32 < K) { stage_async(buf ^ 1, k0 + 32); stage_tdm(buf ^ 1, k0 + 32); }

    v16h af[4], bf[2];
#pragma unroll
    for (int i = 0; i < 4; ++i)
      af[i] = load_a_frag(&sA[buf][(wr + i * 16) * 40], 40);
#pragma unroll
    for (int j = 0; j < 2; ++j)
      bf[j] = load_b_frag(&sB[buf][(wc + j * 16) * 40], 40);
#pragma unroll
    for (int i = 0; i < 4; ++i)
#pragma unroll
      for (int j = 0; j < 2; ++j) acc[i][j] = wmma_f16(af[i], bf[j], acc[i][j]);

    asm volatile("s_wait_asynccnt 0x0" ::: "memory");
    __builtin_amdgcn_s_wait_tensorcnt(0);
    __syncthreads();
  }

  // epilogue: C/D layout -> lane(0..15) col=lane rows 0..7; lane(16..31) rows 8..15
  int r0 = (lane >> 4) * 8;
  int cc = lane & 15;
#pragma unroll
  for (int i = 0; i < 4; ++i)
#pragma unroll
    for (int j = 0; j < 2; ++j) {
      int nIdx = n0 + wc + j * 16 + cc;
      float bval = bias[nIdx];
#pragma unroll
      for (int rr = 0; rr < 8; ++rr) {
        int m = m0 + wr + i * 16 + r0 + rr;
        float v = acc[i][j][rr] + bval;
        if (MODE == 2) {
          ((float*)out)[(size_t)m * 1024 + nIdx] = v;
        } else {
          int b = m >> 11, s = m & 2047, h = nIdx >> 6, d = nIdx & 63;
          size_t dst = (MODE == 0)
              ? (((size_t)(b * 16 + h)) * 2048 + s) * 64 + d
              : (((size_t)(b * 16 + h)) * 64 + d) * 2048 + s;
          ((_Float16*)out)[dst] = (_Float16)v;
        }
      }
    }
}

// ---------------------------------------------------------------------------
// Fused attention: one WG per (b, h, 16-query tile). 8 waves x 256 keys each.
// Full 16x2048 score strip in dynamic LDS (CDNA5: 320KB/WGP).
// scores(WMMA) -> softmax(LDS) -> single HBM write of attn -> ctx(WMMA).
// ---------------------------------------------------------------------------
#define SSTR 2052                       // padded score row stride (floats)
#define SMEM_SCORES (16 * SSTR * 4)     // 131328 B
#define SMEM_Q      (16 * 64 * 2)       // 2048 B
#define SMEM_RED    (256 * 4)           // 1024 B
#define SMEM_ROW    (32 * 4)            // 128 B
#define SMEM_CTX    (8 * 16 * 64 * 4)   // 32768 B
#define SMEM_ATTN   (SMEM_SCORES + SMEM_Q + SMEM_RED + SMEM_ROW + SMEM_CTX)

__global__ __launch_bounds__(256) void attn_kernel(
    const _Float16* __restrict__ qh,   // [b][h][s][64]  (gate+scale folded)
    const _Float16* __restrict__ kh,   // [b][h][s][64]  (ent folded)
    const _Float16* __restrict__ vt,   // [b][h][64][s]
    float* __restrict__ attn_out,      // [b][h][s][s] fp32
    _Float16* __restrict__ ctxh) {     // [tok][1024] f16
  extern __shared__ char smem[];
  float*    sS   = (float*)smem;
  _Float16* sQ   = (_Float16*)(smem + SMEM_SCORES);
  float*    sRed = (float*)(smem + SMEM_SCORES + SMEM_Q);
  float*    sRow = (float*)(smem + SMEM_SCORES + SMEM_Q + SMEM_RED);
  float*    sCtx = (float*)(smem + SMEM_SCORES + SMEM_Q + SMEM_RED + SMEM_ROW);

  int t = threadIdx.x, w = t >> 5, lane = t & 31;
  int idx = blockIdx.x;
  int qt = idx & 127, h = (idx >> 7) & 15, b = idx >> 11;
  int q0 = qt * 16;
  size_t head = (size_t)(b * 16 + h);
  size_t headQK = head * 2048 * 64;
  size_t headV  = head * 64 * 2048;

  {  // stage the 16x64 Q tile
    int r = t >> 4, c = (t & 15) * 4;
    *(v4h*)&sQ[r * 64 + c] = *(const v4h*)(qh + headQK + (size_t)(q0 + r) * 64 + c);
  }
  __syncthreads();

  v16h aq0 = load_a_frag(sQ, 64);        // d = 0..31
  v16h aq1 = load_a_frag(sQ + 32, 64);   // d = 32..63

  // ---- scores: each wave covers 256 keys (16 key-tiles) ----
  for (int kt = 0; kt < 16; ++kt) {
    int n0 = w * 256 + kt * 16;
    const _Float16* kb = kh + headQK + (size_t)n0 * 64;
    v16h b0 = load_b_frag(kb, 64);
    v16h b1 = load_b_frag(kb + 32, 64);
    v8f c = {};
    c = wmma_f16(aq0, b0, c);
    c = wmma_f16(aq1, b1, c);
    int col = n0 + (lane & 15);
    int r0 = (lane >> 4) * 8;
#pragma unroll
    for (int r = 0; r < 8; ++r) sS[(r0 + r) * SSTR + col] = c[r];
  }
  __syncthreads();

  // ---- softmax over 2048 keys (16 threads per row) ----
  {
    int r = t & 15, seg = t >> 4;
    const float* rp = sS + r * SSTR + seg * 128;
    float m = -3.0e38f;
    for (int j = 0; j < 128; ++j) m = fmaxf(m, rp[j]);
    sRed[t] = m;
  }
  __syncthreads();
  if (t < 16) {
    float m = -3.0e38f;
    for (int s = 0; s < 16; ++s) m = fmaxf(m, sRed[s * 16 + t]);
    sRow[t] = m;
  }
  __syncthreads();
  {
    int r = t & 15, seg = t >> 4;
    float m = sRow[r], sum = 0.0f;
    float* rp = sS + r * SSTR + seg * 128;
    for (int j = 0; j < 128; ++j) { float e = __expf(rp[j] - m); rp[j] = e; sum += e; }
    sRed[t] = sum;
  }
  __syncthreads();
  if (t < 16) {
    float s = 0.0f;
    for (int k = 0; k < 16; ++k) s += sRed[k * 16 + t];
    sRow[16 + t] = 1.0f / s;
  }
  __syncthreads();
  {  // normalize in LDS + single coalesced HBM write of attn
    float* aout = attn_out + (head * 2048 + q0) * 2048;
    for (int j = 0; j < 128; ++j) {
      int e = j * 256 + t;                  // 16*2048 elements, coalesced
      int r = e >> 11, c = e & 2047;
      float v = sS[r * SSTR + c] * sRow[16 + r];
      sS[r * SSTR + c] = v;
      aout[(size_t)r * 2048 + c] = v;
    }
  }
  __syncthreads();

  // ---- ctx = attn @ V : wave accumulates its 256 keys, 4 n-tiles of 16 ----
  v8f acc[4] = {};
  for (int s8 = 0; s8 < 8; ++s8) {
    int kb0 = w * 256 + s8 * 32;
    v16h a;
    {
      int row = lane & 15;
      const float* p = sS + row * SSTR + kb0 + ((lane >> 4) << 3);
#pragma unroll
      for (int i = 0; i < 8; ++i) { a[i] = (_Float16)p[i]; a[i + 8] = (_Float16)p[i + 16]; }
    }
#pragma unroll
    for (int ni = 0; ni < 4; ++ni) {
      const _Float16* vb = vt + headV + (size_t)(ni * 16) * 2048 + kb0;
      v16h bb = load_b_frag(vb, 2048);
      acc[ni] = wmma_f16(a, bb, acc[ni]);
    }
  }
  {  // per-wave partials -> LDS
    int r0 = (lane >> 4) * 8, cc = lane & 15;
    float* my = sCtx + w * 1024;
#pragma unroll
    for (int ni = 0; ni < 4; ++ni)
#pragma unroll
      for (int r = 0; r < 8; ++r) my[(r0 + r) * 64 + ni * 16 + cc] = acc[ni][r];
  }
  __syncthreads();
  // cross-wave reduce + write ctx f16 [tok][1024]
  for (int j = 0; j < 4; ++j) {
    int e = j * 256 + t;                    // 1024 outputs
    int r = e >> 6, c = e & 63;
    float s = 0.0f;
#pragma unroll
    for (int wv = 0; wv < 8; ++wv) s += sCtx[wv * 1024 + r * 64 + c];
    ctxh[((size_t)(b * 2048 + q0 + r)) * 1024 + h * 64 + c] = (_Float16)s;
  }
}

// ---------------------------------------------------------------------------
// Host side
// ---------------------------------------------------------------------------
#define OFF_XH   ((size_t)0)          // 4096x1024 f16  (8 MB)
#define OFF_WQ   ((size_t)8388608)    // 1024x1024 f16  (2 MB)
#define OFF_WK   ((size_t)10485760)
#define OFF_WV   ((size_t)12582912)
#define OFF_WO   ((size_t)14680064)
#define OFF_Q    ((size_t)16777216)   // [b][h][s][64] f16 (8 MB)
#define OFF_K    ((size_t)25165824)
#define OFF_VT   ((size_t)33554432)   // [b][h][64][s] f16 (8 MB)
#define OFF_CTX  ((size_t)41943040)   // [tok][1024] f16 (8 MB)
#define OFF_BQ   ((size_t)50331648)   // 1024 f32
#define OFF_BK   ((size_t)50335744)   // 1024 f32

extern "C" void kernel_launch(void* const* d_in, const int* in_sizes, int n_in,
                              void* d_out, int out_size, void* d_ws, size_t ws_size,
                              hipStream_t stream) {
  (void)in_sizes; (void)n_in; (void)out_size; (void)ws_size;
  const float* x    = (const float*)d_in[0];
  const float* Wq   = (const float*)d_in[1];
  const float* bq   = (const float*)d_in[2];
  const float* Wk   = (const float*)d_in[3];
  const float* bk   = (const float*)d_in[4];
  const float* Wv   = (const float*)d_in[5];
  const float* bv   = (const float*)d_in[6];
  const float* Wo   = (const float*)d_in[7];
  const float* bo   = (const float*)d_in[8];
  const float* gate = (const float*)d_in[9];
  const float* ent  = (const float*)d_in[10];

  char* ws = (char*)d_ws;
  _Float16* xh   = (_Float16*)(ws + OFF_XH);
  _Float16* wqh  = (_Float16*)(ws + OFF_WQ);
  _Float16* wkh  = (_Float16*)(ws + OFF_WK);
  _Float16* wvh  = (_Float16*)(ws + OFF_WV);
  _Float16* woh  = (_Float16*)(ws + OFF_WO);
  _Float16* qh   = (_Float16*)(ws + OFF_Q);
  _Float16* kh   = (_Float16*)(ws + OFF_K);
  _Float16* vth  = (_Float16*)(ws + OFF_VT);
  _Float16* ctxh = (_Float16*)(ws + OFF_CTX);
  float*    bqe  = (float*)(ws + OFF_BQ);
  float*    bke  = (float*)(ws + OFF_BK);

  // 1) conversions + weight folding
  cvt_f16<<<16384, 256, 0, stream>>>(x, xh, NTOK * D_MODEL);
  cvt_f16<<<4096, 256, 0, stream>>>(Wv, wvh, D_MODEL * D_MODEL);
  cvt_f16<<<4096, 256, 0, stream>>>(Wo, woh, D_MODEL * D_MODEL);
  prep_wq<<<4096, 256, 0, stream>>>(Wq, gate, wqh);
  prep_wk<<<4096, 256, 0, stream>>>(Wk, ent, wkh);
  prep_bq<<<4, 256, 0, stream>>>(bq, gate, bqe);
  prep_bk<<<4, 256, 0, stream>>>(bk, ent, bke);

  // 2) QKV projections (WMMA GEMMs, fused gate/scale/entanglement)
  dim3 gg(8, 32);  // N/128, M/128
  gemm_wmma<0><<<gg, 256, 0, stream>>>(xh, wqh, bqe, (void*)qh);
  gemm_wmma<0><<<gg, 256, 0, stream>>>(xh, wkh, bke, (void*)kh);
  gemm_wmma<1><<<gg, 256, 0, stream>>>(xh, wvh, bv, (void*)vth);

  // 3) fused attention (scores + softmax + attn write + ctx), big dynamic LDS
  float* attn_out = (float*)d_out + (size_t)NTOK * D_MODEL;  // after `out`
  hipFuncSetAttribute((const void*)attn_kernel,
                      hipFuncAttributeMaxDynamicSharedMemorySize, SMEM_ATTN);
  attn_kernel<<<4096, 256, SMEM_ATTN, stream>>>(qh, kh, vth, attn_out, ctxh);

  // 4) output projection -> fp32 out
  gemm_wmma<2><<<gg, 256, 0, stream>>>(ctxh, woh, bo, d_out);
}
